// SparseMoeBlock_68874095558903
// MI455X (gfx1250) — matrix operand order repaired and checked by reading, not compiled, not run on hardware
//
#include <hip/hip_runtime.h>
#include <hip/hip_bf16.h>

// Problem constants (fixed by the reference setup).
#define T_TOK 4096   // B*S tokens
#define HDIM  2048
#define IDIM  2816
#define NEXP  8

// GEMM tiling: 64x128 block tile, K-step 32 (one bf16 WMMA K), 8 waves of 32.
#define BM 64
#define BN 128
#define BK 32
#define LDSK 48      // padded LDS row stride in bf16 elems (96B: 16B-aligned, spreads banks)

typedef __attribute__((ext_vector_type(16))) __bf16 v16bf;
typedef __attribute__((ext_vector_type(8)))  __bf16 bf16x8;
typedef __attribute__((ext_vector_type(8)))  float  v8f;

// The async-to-LDS builtin exists on this toolchain (probe-confirmed), but its
// pointer parameter types are awkward; use the documented instruction directly.
#if __has_builtin(__builtin_amdgcn_global_load_async_to_lds_b128)
#define HAVE_ASYNC_LDS 1
#endif

__device__ __forceinline__ void wait_async0() {
#if __has_builtin(__builtin_amdgcn_s_wait_asynccnt)
  __builtin_amdgcn_s_wait_asynccnt(0);
#else
  asm volatile("s_wait_asynccnt 0x0" ::: "memory");
#endif
}

// Issue GLOBAL_LOAD_ASYNC_TO_LDS_B128: 16 bytes per lane, LDS byte offset in a
// VGPR, 64-bit global address in a VGPR pair. Tracked by ASYNCcnt.
__device__ __forceinline__ void async_copy_b128(const void* gsrc, void* lds_dst) {
  const unsigned lds_off =
      (unsigned)(unsigned long long)(__attribute__((address_space(3))) void*)lds_dst;
  asm volatile("global_load_async_to_lds_b128 %0, %1, off"
               :
               : "v"(lds_off), "v"(gsrc)
               : "memory");
}

__device__ __forceinline__ __bf16 f2bf(float f) {
  union { float f; unsigned u; } v; v.f = f;
  unsigned r = (v.u + 0x7FFFu + ((v.u >> 16) & 1u)) >> 16;   // RNE
  union { unsigned short s; __bf16 b; } o; o.s = (unsigned short)r;
  return o.b;
}

__device__ __forceinline__ v8f vzero8() {
  v8f z;
#pragma unroll
  for (int i = 0; i < 8; ++i) z[i] = 0.f;
  return z;
}

// A fragment 16x32 bf16 (M x K): lanes 0-15 -> M=lane, K=0..15;
// lanes 16-31 -> M=lane-16, K=16..31.
__device__ __forceinline__ v16bf frag_a(const __bf16* lds, int row0, int lane) {
  const int m    = row0 + (lane & 15);
  const int koff = (lane & 16) ? 16 : 0;
  const bf16x8* p = (const bf16x8*)(lds + m * LDSK + koff);
  bf16x8 lo = p[0];      // K = koff .. koff+7
  bf16x8 hi = p[1];      // K = koff+8 .. koff+15
  v16bf f;
#pragma unroll
  for (int i = 0; i < 8; ++i) { f[i] = lo[i]; f[i + 8] = hi[i]; }
  return f;
}

// B fragment 32x16 bf16 (K x N), stored as weight rows (N-major, contiguous K):
// lanes 0-15 -> N=lane, K=0..15; lanes 16-31 -> N=lane-16, K=16..31.
__device__ __forceinline__ v16bf frag_b(const __bf16* lds, int col0, int lane) {
  const int n    = col0 + (lane & 15);
  const int koff = (lane & 16) ? 16 : 0;
  const bf16x8* p = (const bf16x8*)(lds + n * LDSK + koff);
  bf16x8 lo = p[0];
  bf16x8 hi = p[1];
  v16bf f;
#pragma unroll
  for (int i = 0; i < 8; ++i) { f[i] = lo[i]; f[i + 8] = hi[i]; }
  return f;
}

// ---------------------------------------------------------------------------
// Router: logits, top-2 softmax weights, per-expert token lists.
// ---------------------------------------------------------------------------
__global__ __launch_bounds__(256) void moe_router(
    const float* __restrict__ x, const float* __restrict__ gw,
    float* __restrict__ logits, int* __restrict__ counts,
    int* __restrict__ list, float* __restrict__ wlist) {
  const int lane = threadIdx.x & 31;
  const int t = blockIdx.x * 8 + (threadIdx.x >> 5);
  const float* __restrict__ xr = x + (size_t)t * HDIM;

  float acc[NEXP];
#pragma unroll
  for (int e = 0; e < NEXP; ++e) acc[e] = 0.f;
  for (int h = lane; h < HDIM; h += 32) {
    const float xv = xr[h];
#pragma unroll
    for (int e = 0; e < NEXP; ++e) acc[e] = fmaf(xv, gw[e * HDIM + h], acc[e]);
  }
#pragma unroll
  for (int e = 0; e < NEXP; ++e)
    for (int off = 16; off; off >>= 1) acc[e] += __shfl_xor(acc[e], off, 32);

  if (lane == 0) {
    int i0 = 0;
#pragma unroll
    for (int e = 1; e < NEXP; ++e) if (acc[e] > acc[i0]) i0 = e;
    int i1 = (i0 == 0) ? 1 : 0;
#pragma unroll
    for (int e = 0; e < NEXP; ++e) if (e != i0 && acc[e] > acc[i1]) i1 = e;
    // normalized top-2 softmax weights: w0 = 1/(1+exp(l1-l0)), w1 = 1-w0
    const float r  = __expf(acc[i1] - acc[i0]);
    const float w0 = 1.f / (1.f + r);
    const float w1 = r * w0;
#pragma unroll
    for (int e = 0; e < NEXP; ++e) logits[(size_t)t * NEXP + e] = acc[e];
    const int p0 = atomicAdd(&counts[i0], 1);
    list[i0 * T_TOK + p0] = t; wlist[i0 * T_TOK + p0] = w0;
    const int p1 = atomicAdd(&counts[i1], 1);
    list[i1 * T_TOK + p1] = t; wlist[i1 * T_TOK + p1] = w1;
  }
}

__global__ void moe_offsets(const int* __restrict__ counts, int* __restrict__ base) {
  if (threadIdx.x == 0) {
    int s = 0;
    for (int e = 0; e < NEXP; ++e) { base[e] = s; s += counts[e]; }
  }
}

// ---------------------------------------------------------------------------
// GEMM1: for expert e, gathered tokens -> h = silu(x@Gᵀ) * (x@Uᵀ), bf16 out.
// ---------------------------------------------------------------------------
__global__ __launch_bounds__(256) void moe_gemm1(
    const float* __restrict__ x, const float* __restrict__ gatep,
    const float* __restrict__ upp, const int* __restrict__ counts,
    const int* __restrict__ base, const int* __restrict__ list,
    __bf16* __restrict__ hbuf) {
  const int e  = blockIdx.z;
  const int ne = counts[e];
  const int m0 = blockIdx.y * BM;
  if (m0 >= ne) return;
  const int n0 = blockIdx.x * BN;

  __shared__ __align__(16) __bf16 lA[BM * LDSK];
  __shared__ __align__(16) __bf16 lG[BN * LDSK];
  __shared__ __align__(16) __bf16 lU[BN * LDSK];

  const int tid  = threadIdx.x;
  const int lane = tid & 31;

  // A loader: 64 rows x 32 cols fp32, rows gathered by token id; out-of-range
  // rows clamp to the last valid row (their C rows are discarded) -> no divergence.
  const int ar = tid >> 2;
  const int ac = (tid & 3) * 8;
  const int arow = (m0 + ar < ne) ? (m0 + ar) : (ne - 1);
  const float* aptr = x + (size_t)list[e * T_TOK + arow] * HDIM;

  // B loader: 128 rows x 32 cols fp32, 16 floats/thread per weight matrix.
  const int br = tid >> 1;
  const int bc = (tid & 1) * 16;
  const float* gptr = gatep + ((size_t)e * IDIM + (n0 + br)) * HDIM;
  const float* uptr = upp   + ((size_t)e * IDIM + (n0 + br)) * HDIM;

  // Wave grid 2x4 over the 64x128 tile; each wave owns a 32x32 sub-tile.
  const int mb = ((tid >> 5) & 1) * 32;
  const int nb = (tid >> 6) * 32;

  v8f accg[2][2], accu[2][2];
#pragma unroll
  for (int i = 0; i < 2; ++i)
#pragma unroll
    for (int j = 0; j < 2; ++j) { accg[i][j] = vzero8(); accu[i][j] = vzero8(); }

  for (int k0 = 0; k0 < HDIM; k0 += BK) {
    float av[8];
    {
      const float4* p = (const float4*)(aptr + k0 + ac);
      float4 v0 = p[0], v1 = p[1];
      av[0] = v0.x; av[1] = v0.y; av[2] = v0.z; av[3] = v0.w;
      av[4] = v1.x; av[5] = v1.y; av[6] = v1.z; av[7] = v1.w;
    }
    float gvv[16], uvv[16];
    {
      const float4* pg = (const float4*)(gptr + k0 + bc);
      const float4* pu = (const float4*)(uptr + k0 + bc);
#pragma unroll
      for (int q = 0; q < 4; ++q) {
        float4 g4 = pg[q], u4 = pu[q];
        gvv[q * 4 + 0] = g4.x; gvv[q * 4 + 1] = g4.y; gvv[q * 4 + 2] = g4.z; gvv[q * 4 + 3] = g4.w;
        uvv[q * 4 + 0] = u4.x; uvv[q * 4 + 1] = u4.y; uvv[q * 4 + 2] = u4.z; uvv[q * 4 + 3] = u4.w;
      }
    }
    if (k0 + BK < HDIM) {  // WGP-scope prefetch of next weight cachelines
      __builtin_prefetch(gptr + k0 + BK + bc, 0, 3);
      __builtin_prefetch(uptr + k0 + BK + bc, 0, 3);
    }
    __syncthreads();
#pragma unroll
    for (int j = 0; j < 8; ++j) lA[ar * LDSK + ac + j] = f2bf(av[j]);
#pragma unroll
    for (int j = 0; j < 16; ++j) {
      lG[br * LDSK + bc + j] = f2bf(gvv[j]);
      lU[br * LDSK + bc + j] = f2bf(uvv[j]);
    }
    __syncthreads();

    // Hoist all fragment loads so the ds_load_b128s clause and overlap the WMMAs.
    const v16bf a0  = frag_a(lA, mb, lane);
    const v16bf a1  = frag_a(lA, mb + 16, lane);
    const v16bf bg0 = frag_b(lG, nb, lane);
    const v16bf bu0 = frag_b(lU, nb, lane);
    const v16bf bg1 = frag_b(lG, nb + 16, lane);
    const v16bf bu1 = frag_b(lU, nb + 16, lane);
    accg[0][0] = __builtin_amdgcn_wmma_f32_16x16x32_bf16(false, a0, false, bg0, (short)0, accg[0][0], false, false);
    accg[1][0] = __builtin_amdgcn_wmma_f32_16x16x32_bf16(false, a1, false, bg0, (short)0, accg[1][0], false, false);
    accu[0][0] = __builtin_amdgcn_wmma_f32_16x16x32_bf16(false, a0, false, bu0, (short)0, accu[0][0], false, false);
    accu[1][0] = __builtin_amdgcn_wmma_f32_16x16x32_bf16(false, a1, false, bu0, (short)0, accu[1][0], false, false);
    accg[0][1] = __builtin_amdgcn_wmma_f32_16x16x32_bf16(false, a0, false, bg1, (short)0, accg[0][1], false, false);
    accg[1][1] = __builtin_amdgcn_wmma_f32_16x16x32_bf16(false, a1, false, bg1, (short)0, accg[1][1], false, false);
    accu[0][1] = __builtin_amdgcn_wmma_f32_16x16x32_bf16(false, a0, false, bu1, (short)0, accu[0][1], false, false);
    accu[1][1] = __builtin_amdgcn_wmma_f32_16x16x32_bf16(false, a1, false, bu1, (short)0, accu[1][1], false, false);
  }

  // SwiGLU epilogue -> packed per-expert bf16 activation buffer.
  const int hb    = base[e];
  const int nlane = lane & 15;
  const int mhalf = (lane >> 4) * 8;
#pragma unroll
  for (int mf = 0; mf < 2; ++mf)
#pragma unroll
    for (int nf = 0; nf < 2; ++nf)
#pragma unroll
      for (int r = 0; r < 8; ++r) {
        const int mrow = m0 + mb + mf * 16 + mhalf + r;
        if (mrow < ne) {
          const float g = accg[mf][nf][r];
          const float u = accu[mf][nf][r];
          const float hval = (g / (1.f + __expf(-g))) * u;
          hbuf[(size_t)(hb + mrow) * IDIM + (n0 + nb + nf * 16 + nlane)] = f2bf(hval);
        }
      }
}

// ---------------------------------------------------------------------------
// GEMM2: out[token] += w * (h @ Dᵀ).  A is already bf16 -> stream it straight
// into LDS with GLOBAL_LOAD_ASYNC_TO_LDS_B128 (ASYNCcnt). Scatter via f32
// atomics (each out element gets exactly top_k=2 commutative adds).
// ---------------------------------------------------------------------------
__global__ __launch_bounds__(256) void moe_gemm2(
    const __bf16* __restrict__ hbuf, const float* __restrict__ downp,
    const int* __restrict__ counts, const int* __restrict__ base,
    const int* __restrict__ list, const float* __restrict__ wlist,
    float* __restrict__ out) {
  const int e  = blockIdx.z;
  const int ne = counts[e];
  const int m0 = blockIdx.y * BM;
  if (m0 >= ne) return;
  const int n0 = blockIdx.x * BN;

  __shared__ __align__(16) __bf16 lA[BM * LDSK];
  __shared__ __align__(16) __bf16 lB[BN * LDSK];

  const int tid  = threadIdx.x;
  const int lane = tid & 31;
  const int ar = tid >> 2;
  const int ac = (tid & 3) * 8;
  const int arow = (m0 + ar < ne) ? (m0 + ar) : (ne - 1);   // clamp, rows >= ne discarded
  const __bf16* aptr = hbuf + (size_t)(base[e] + arow) * IDIM;
  const int br = tid >> 1;
  const int bc = (tid & 1) * 16;
  const float* bptr = downp + ((size_t)e * HDIM + (n0 + br)) * IDIM;

  const int mb = ((tid >> 5) & 1) * 32;
  const int nb = (tid >> 6) * 32;

  v8f acc[2][2];
#pragma unroll
  for (int i = 0; i < 2; ++i)
#pragma unroll
    for (int j = 0; j < 2; ++j) acc[i][j] = vzero8();

  for (int k0 = 0; k0 < IDIM; k0 += BK) {
#ifndef HAVE_ASYNC_LDS
    const bf16x8 avec = *(const bf16x8*)(aptr + k0 + ac);
#endif
    float bvv[16];
    {
      const float4* pb = (const float4*)(bptr + k0 + bc);
#pragma unroll
      for (int q = 0; q < 4; ++q) {
        float4 b4 = pb[q];
        bvv[q * 4 + 0] = b4.x; bvv[q * 4 + 1] = b4.y; bvv[q * 4 + 2] = b4.z; bvv[q * 4 + 3] = b4.w;
      }
    }
    if (k0 + BK < IDIM) __builtin_prefetch(bptr + k0 + BK + bc, 0, 3);
    __syncthreads();   // prior iteration's readers are done with LDS
#ifdef HAVE_ASYNC_LDS
    async_copy_b128(aptr + k0 + ac, &lA[ar * LDSK + ac]);
#else
    *(bf16x8*)(&lA[ar * LDSK + ac]) = avec;
#endif
#pragma unroll
    for (int j = 0; j < 16; ++j) lB[br * LDSK + bc + j] = f2bf(bvv[j]);
#ifdef HAVE_ASYNC_LDS
    wait_async0();     // our async tile portion is in LDS before we signal
#endif
    __syncthreads();

    const v16bf a0 = frag_a(lA, mb, lane);
    const v16bf a1 = frag_a(lA, mb + 16, lane);
    const v16bf b0 = frag_b(lB, nb, lane);
    const v16bf b1 = frag_b(lB, nb + 16, lane);
    acc[0][0] = __builtin_amdgcn_wmma_f32_16x16x32_bf16(false, a0, false, b0, (short)0, acc[0][0], false, false);
    acc[1][0] = __builtin_amdgcn_wmma_f32_16x16x32_bf16(false, a1, false, b0, (short)0, acc[1][0], false, false);
    acc[0][1] = __builtin_amdgcn_wmma_f32_16x16x32_bf16(false, a0, false, b1, (short)0, acc[0][1], false, false);
    acc[1][1] = __builtin_amdgcn_wmma_f32_16x16x32_bf16(false, a1, false, b1, (short)0, acc[1][1], false, false);
  }

  const int nlane = lane & 15;
  const int mhalf = (lane >> 4) * 8;
#pragma unroll
  for (int mf = 0; mf < 2; ++mf)
#pragma unroll
    for (int nf = 0; nf < 2; ++nf)
#pragma unroll
      for (int r = 0; r < 8; ++r) {
        const int mrow = m0 + mb + mf * 16 + mhalf + r;
        if (mrow < ne) {
          const int   t = list[e * T_TOK + mrow];
          const float w = wlist[e * T_TOK + mrow];
          unsafeAtomicAdd(&out[(size_t)t * HDIM + (n0 + nb + nf * 16 + nlane)],
                          acc[mf][nf][r] * w);
        }
      }
}

// ---------------------------------------------------------------------------
extern "C" void kernel_launch(void* const* d_in, const int* in_sizes, int n_in,
                              void* d_out, int out_size, void* d_ws, size_t ws_size,
                              hipStream_t stream) {
  const float* x  = (const float*)d_in[0];  // [T,H]
  const float* gw = (const float*)d_in[1];  // [E,H]
  const float* gp = (const float*)d_in[2];  // [E,I,H]
  const float* up = (const float*)d_in[3];  // [E,I,H]
  const float* dp = (const float*)d_in[4];  // [E,H,I]
  // d_in[5] = top_k (fixed 2)

  float* out    = (float*)d_out;                       // [T,H]
  float* logits = out + (size_t)T_TOK * HDIM;          // [T,E]

  // Workspace layout
  char*   ws     = (char*)d_ws;
  int*    counts = (int*)ws;                                            // 8 ints
  int*    base   = counts + 8;                                          // 8 ints
  int*    list   = (int*)(ws + 256);                                    // [E][T]
  float*  wlist  = (float*)(ws + 256 + sizeof(int) * NEXP * T_TOK);     // [E][T]
  __bf16* hbuf   = (__bf16*)(ws + 256 + 2 * sizeof(int) * NEXP * T_TOK);
  // hbuf: (2*T + BM) rows x IDIM bf16 ~= 44.4 MiB (16B aligned offset)
  (void)in_sizes; (void)n_in; (void)out_size; (void)ws_size;

  (void)hipMemsetAsync(counts, 0, sizeof(int) * 16, stream);
  (void)hipMemsetAsync(out, 0, sizeof(float) * (size_t)T_TOK * HDIM, stream);

  moe_router<<<T_TOK / 8, 256, 0, stream>>>(x, gw, logits, counts, list, wlist);
  moe_offsets<<<1, 32, 0, stream>>>(counts, base);
  moe_gemm1<<<dim3(IDIM / BN, T_TOK / BM, NEXP), 256, 0, stream>>>(
      x, gp, up, counts, base, list, hbuf);
  moe_gemm2<<<dim3(HDIM / BN, T_TOK / BM, NEXP), 256, 0, stream>>>(
      hbuf, dp, counts, base, list, wlist, out);
}
